// LossComputation_807453852256
// MI455X (gfx1250) — matrix operand order repaired and checked by reading, not compiled, not run on hardware
//
#include <hip/hip_runtime.h>
#include <hip/hip_bf16.h>
#include <math.h>

typedef __attribute__((ext_vector_type(16))) __bf16 v16bf;
typedef __attribute__((ext_vector_type(8)))  __bf16 v8bf;
typedef __attribute__((ext_vector_type(8)))  float  v8f;
typedef __attribute__((ext_vector_type(4)))  unsigned int u32x4;
typedef __attribute__((ext_vector_type(8)))  int i32x8;
typedef __attribute__((ext_vector_type(4)))  int i32x4;

#define B_SZ 1024
#define D_SZ 1024
#define C_SZ 11003
#define C_PAD 11008              // next multiple of 32/128
#define SHIFT_INST 28.0f
#define SHIFT_CMPC 20.0f

// ---------------- block reduction helpers (blockDim must be power of 2) ----------------
__device__ __forceinline__ float block_sum(float v, float* sh) {
  int t = threadIdx.x;
  sh[t] = v; __syncthreads();
  for (int s = blockDim.x >> 1; s > 0; s >>= 1) {
    if (t < s) sh[t] += sh[t + s];
    __syncthreads();
  }
  float r = sh[0]; __syncthreads();
  return r;
}
__device__ __forceinline__ float block_max(float v, float* sh) {
  int t = threadIdx.x;
  sh[t] = v; __syncthreads();
  for (int s = blockDim.x >> 1; s > 0; s >>= 1) {
    if (t < s) sh[t] = fmaxf(sh[t], sh[t + s]);
    __syncthreads();
  }
  float r = sh[0]; __syncthreads();
  return r;
}

// ---------------- init ----------------
__global__ void zero_u32_kernel(unsigned int* p, int n) {
  int i = blockIdx.x * blockDim.x + threadIdx.x;
  if (i < n) p[i] = 0u;
}

// ------- fused transpose + f32->bf16 convert + column sum-of-squares -------------------
// In : W  (D x C, f32, row-major)
// Out: Wt (C_PAD x D, bf16, row-major) ; rows >= C_SZ are zero-filled
//      cnsq[c] += sum_d W[d,c]^2   (cnsq pre-zeroed, size C_PAD)
__global__ void convT_kernel(const float* __restrict__ W, __bf16* __restrict__ Wt,
                             float* __restrict__ cnsq) {
  __shared__ __bf16 tile[32][33];
  __shared__ float csum[32];
  int c0 = blockIdx.x * 32;
  int k0 = blockIdx.y * 32;
  int tc = threadIdx.x & 31;
  int tr = threadIdx.x >> 5;                 // 0..7
  if (threadIdx.x < 32) csum[threadIdx.x] = 0.f;
  __syncthreads();
  int c = c0 + tc;
  bool cvld = (c < C_SZ);
  float sq = 0.f;
#pragma unroll
  for (int j = 0; j < 4; ++j) {
    int kl = tr + 8 * j;
    float x = cvld ? W[(size_t)(k0 + kl) * C_SZ + c] : 0.f;
    sq = fmaf(x, x, sq);
    tile[kl][tc] = (__bf16)x;                // tile[k_local][c_local]
  }
  atomicAdd(&csum[tc], sq);
  __syncthreads();
#pragma unroll
  for (int j = 0; j < 4; ++j) {
    int cl = tr + 8 * j;                     // c_local on write side
    Wt[(size_t)(c0 + cl) * D_SZ + (k0 + tc)] = tile[tc][cl];
  }
  if (threadIdx.x < 32) atomicAdd(&cnsq[c0 + threadIdx.x], csum[threadIdx.x]);
}

// ---------------- row-normalize embeddings, make bf16 copies, cmpc row coefficients ----
__global__ void prep_kernel(const float* __restrict__ V, const float* __restrict__ T,
                            __bf16* __restrict__ vnb, __bf16* __restrict__ tnb,
                            __bf16* __restrict__ vb,  __bf16* __restrict__ tb,
                            float* __restrict__ arow, float* __restrict__ brow) {
  __shared__ float sh[256];
  int i = blockIdx.x;
  const float* vr = V + (size_t)i * D_SZ;
  const float* tr = T + (size_t)i * D_SZ;
  float vv = 0.f, tt = 0.f, vt = 0.f;
  for (int d = threadIdx.x; d < D_SZ; d += blockDim.x) {
    float v = vr[d], t = tr[d];
    vv = fmaf(v, v, vv); tt = fmaf(t, t, tt); vt = fmaf(v, t, vt);
  }
  vv = block_sum(vv, sh);
  tt = block_sum(tt, sh);
  vt = block_sum(vt, sh);
  float vi = rsqrtf(vv), ti = rsqrtf(tt);
  for (int d = threadIdx.x; d < D_SZ; d += blockDim.x) {
    float v = vr[d], t = tr[d];
    vnb[(size_t)i * D_SZ + d] = (__bf16)(v * vi);
    tnb[(size_t)i * D_SZ + d] = (__bf16)(t * ti);
    vb [(size_t)i * D_SZ + d] = (__bf16)v;
    tb [(size_t)i * D_SZ + d] = (__bf16)t;
  }
  if (threadIdx.x == 0) { arow[i] = vt * ti; brow[i] = vt * vi; }
}

// ---------------- B x B GEMM: out = A @ Bm^T  (bf16 in, f32 out) ----------------
__global__ void gemm_nt_kernel(const __bf16* __restrict__ A, const __bf16* __restrict__ Bm,
                               float* __restrict__ out) {
  int wave = threadIdx.x >> 5;
  int lane = threadIdx.x & 31;
  int half = lane >> 4;
  int lm   = lane & 15;
  int m0 = blockIdx.y * 16;
  int j0 = (blockIdx.x * 8 + wave) * 16;
  const __bf16* ar = A  + (size_t)(m0 + lm) * D_SZ + 8 * half;
  const __bf16* br = Bm + (size_t)(j0 + lm) * D_SZ + 8 * half;
  v8f acc = {0.f,0.f,0.f,0.f,0.f,0.f,0.f,0.f};
  for (int k0 = 0; k0 < D_SZ; k0 += 32) {
    __builtin_prefetch((const void*)(ar + k0 + 128), 0, 0);
    __builtin_prefetch((const void*)(br + k0 + 128), 0, 0);
    v8bf alo = *(const v8bf*)(ar + k0);
    v8bf ahi = *(const v8bf*)(ar + k0 + 16);
    v16bf a = __builtin_shufflevector(alo, ahi, 0,1,2,3,4,5,6,7,8,9,10,11,12,13,14,15);
    v8bf blo = *(const v8bf*)(br + k0);
    v8bf bhi = *(const v8bf*)(br + k0 + 16);
    v16bf b = __builtin_shufflevector(blo, bhi, 0,1,2,3,4,5,6,7,8,9,10,11,12,13,14,15);
    acc = __builtin_amdgcn_wmma_f32_16x16x32_bf16(false, a, false, b, (short)0, acc,
                                                  false, false);
  }
#pragma unroll
  for (int r = 0; r < 8; ++r) {
    int m = m0 + r + 8 * half;
    out[(size_t)m * B_SZ + (j0 + lm)] = acc[r];
  }
}

// ------- fused GEMM (B x D) @ Wt^T (C x D bf16) + online cross-entropy epilogue --------
// The 16 x D A-block (32KB, contiguous rows) is staged into LDS once per block:
// via the Tensor Data Mover when available (wave 0 issues TENSOR_LOAD_TO_LDS, waits on
// TENSORcnt), else via a cooperative vector copy. All 8 waves then read A from LDS.
__global__ void gemm_ce_kernel(const __bf16* __restrict__ A, const __bf16* __restrict__ Wt,
                               const float* __restrict__ cnsq,
                               const float* __restrict__ rowscale, float rs_const,
                               float shift,
                               float* __restrict__ sumexp, float* __restrict__ lablog,
                               unsigned long long* __restrict__ argkey,
                               const int* __restrict__ labels) {
  __shared__ __bf16 As[16 * D_SZ];           // 32 KB A-block
  __shared__ float rowsum[16];
  __shared__ unsigned long long rowkey[16];
  int wave = threadIdx.x >> 5;
  int lane = threadIdx.x & 31;
  int half = lane >> 4;
  int lm   = lane & 15;
  int m0 = blockIdx.y * 16;
  int c0 = (blockIdx.x * 8 + wave) * 16;
  if (threadIdx.x < 16) { rowsum[threadIdx.x] = 0.f; rowkey[threadIdx.x] = 0ull; }

#if defined(__gfx1250__) && __has_builtin(__builtin_amdgcn_tensor_load_to_lds)
  if (threadIdx.x < 32) {                    // wave 0 issues the DMA once
    unsigned lds_addr = (unsigned)(size_t)(&As[0]);       // LDS byte offset (low 32 bits)
    unsigned long long ga = (unsigned long long)(size_t)(A + (size_t)m0 * D_SZ);
    // D# group 0: count=1 | lds_addr | global_addr[56:0] | type=2
    u32x4 g0 = { 1u, lds_addr,
                 (unsigned)(ga & 0xffffffffull),
                 (unsigned)((ga >> 32) & 0x01ffffffull) | 0x80000000u };
    // D# group 1: data_size=2B; tensor 1024 x 16; tile 1024 x 16; dim0 stride 1024
    i32x8 g1 = { (int)(1u << 16),                               // data_size=1 (2 bytes)
                 (int)((unsigned)(D_SZ & 0xffff) << 16),        // tensor_dim0 lo16
                 (int)((D_SZ >> 16) | (16u << 16)),             // dim0 hi | tensor_dim1 lo
                 (int)((unsigned)D_SZ << 16),                   // tile_dim0
                 (int)16,                                       // tile_dim1 (| tile_dim2<<16)
                 (int)D_SZ,                                     // tensor_dim0_stride lo32
                 0, 0 };
    i32x4 z4 = { 0, 0, 0, 0 };
#if __clang_major__ >= 23
    i32x8 z8 = { 0, 0, 0, 0, 0, 0, 0, 0 };
    __builtin_amdgcn_tensor_load_to_lds(g0, g1, z4, z4, z8, 0);
#else
    __builtin_amdgcn_tensor_load_to_lds(g0, g1, z4, z4, 0);
#endif
    __builtin_amdgcn_s_wait_tensorcnt(0);
  }
#else
  {
    const v8bf* src = (const v8bf*)(A + (size_t)m0 * D_SZ);
    v8bf* dst = (v8bf*)As;
    for (int idx = threadIdx.x; idx < 16 * D_SZ / 8; idx += blockDim.x) dst[idx] = src[idx];
  }
#endif
  __syncthreads();

  int  c  = c0 + lm;                          // always < C_PAD; masked vs C_SZ below
  bool cv = (c < C_SZ);
  const __bf16* br = Wt + (size_t)c * D_SZ + 8 * half;
  int aoff = lm * D_SZ + 8 * half;            // A fragment base inside LDS
  v8f acc = {0.f,0.f,0.f,0.f,0.f,0.f,0.f,0.f};
  for (int k0 = 0; k0 < D_SZ; k0 += 32) {
    __builtin_prefetch((const void*)(br + k0 + 128), 0, 0);
    v8bf alo = *(const v8bf*)(&As[aoff + k0]);
    v8bf ahi = *(const v8bf*)(&As[aoff + k0 + 16]);
    v16bf a = __builtin_shufflevector(alo, ahi, 0,1,2,3,4,5,6,7,8,9,10,11,12,13,14,15);
    v8bf blo = *(const v8bf*)(br + k0);
    v8bf bhi = *(const v8bf*)(br + k0 + 16);
    v16bf b = __builtin_shufflevector(blo, bhi, 0,1,2,3,4,5,6,7,8,9,10,11,12,13,14,15);
    acc = __builtin_amdgcn_wmma_f32_16x16x32_bf16(false, a, false, b, (short)0, acc,
                                                  false, false);
  }

  float cninv = rsqrtf(cnsq[c]);
#pragma unroll
  for (int r = 0; r < 8; ++r) {
    int m = m0 + r + 8 * half;
    float rs = rowscale ? rowscale[m] : rs_const;
    float logit = rs * acc[r] * cninv;
    float e = cv ? __expf(logit - shift) : 0.f;
#pragma unroll
    for (int off = 1; off < 16; off <<= 1) e += __shfl_xor(e, off, 32);
    if (lm == 0) atomicAdd(&rowsum[r + 8 * half], e);
    if (cv && labels[m] == c) lablog[m] = logit;
    if (argkey) {
      unsigned int bits = __float_as_uint(logit);
      unsigned int ord  = (bits & 0x80000000u) ? ~bits : (bits | 0x80000000u);
      unsigned long long key = cv ? ((((unsigned long long)ord) << 32) |
                                     (unsigned long long)(unsigned int)c) : 0ull;
#pragma unroll
      for (int off = 1; off < 16; off <<= 1) {
        unsigned long long o = __shfl_xor(key, off, 32);
        key = (o > key) ? o : key;
      }
      if (lm == 0) atomicMax(&rowkey[r + 8 * half], key);
    }
  }
  __syncthreads();
  if (threadIdx.x < 16) {
    atomicAdd(&sumexp[m0 + threadIdx.x], rowsum[threadIdx.x]);
    if (argkey) atomicMax(&argkey[m0 + threadIdx.x], rowkey[threadIdx.x]);
  }
}

// ---------------- label co-occurrence counts ------------------------------------------
__global__ void lcnt_kernel(const int* __restrict__ labels, float* __restrict__ cnt) {
  int i = blockIdx.x * blockDim.x + threadIdx.x;
  if (i >= B_SZ) return;
  int li = labels[i], c = 0;
  for (int j = 0; j < B_SZ; ++j) c += (labels[j] == li) ? 1 : 0;
  cnt[i] = (float)c;
}

// ---------------- global-align sum + MH row-max over sim rows --------------------------
__global__ void ga_mh_row_kernel(const float* __restrict__ sim, const int* __restrict__ labels,
                                 float* __restrict__ ga_accum, float* __restrict__ rowmax_s) {
  __shared__ float sh[256];
  int i = blockIdx.x;
  const float* row = sim + (size_t)i * B_SZ;
  float di = row[i];
  int li = labels[i];
  float ga = 0.f, mx = 0.f;
  for (int j = threadIdx.x; j < B_SZ; j += blockDim.x) {
    float s = row[j];
    bool msk = (labels[j] == li);
    float lp = log1pf(__expf(-10.0f * (s - 0.6f)));
    float ln = log1pf(__expf( 40.0f * (s - 0.4f)));
    ga += msk ? lp : ln;
    float csv = msk ? 0.f : fmaxf(0.2f + s - di, 0.f);
    mx = fmaxf(mx, csv);
  }
  ga = block_sum(ga, sh);
  mx = block_max(mx, sh);
  if (threadIdx.x == 0) { atomicAdd(ga_accum, ga); rowmax_s[i] = mx; }
}

// ---------------- MH column-max of cost_im over sim columns ----------------------------
__global__ void colmax_kernel(const float* __restrict__ sim, const int* __restrict__ labels,
                              float* __restrict__ colmax_im) {
  int j = blockIdx.x * blockDim.x + threadIdx.x;
  if (j >= B_SZ) return;
  float dj = sim[(size_t)j * B_SZ + j];
  int lj = labels[j];
  float mx = 0.f;
  for (int i = 0; i < B_SZ; ++i) {
    float s = sim[(size_t)i * B_SZ + j];
    bool msk = (labels[i] == lj);
    float ci = msk ? 0.f : fmaxf(0.2f + s - dj, 0.f);
    mx = fmaxf(mx, ci);
  }
  colmax_im[j] = mx;
}

// ---------------- CMPM row pass: softmax KL against mask_norm --------------------------
__global__ void cmpm_row_kernel(const float* __restrict__ P, const int* __restrict__ labels,
                                const float* __restrict__ cnt, float* __restrict__ accum) {
  __shared__ float sh[256];
  int i = blockIdx.x;
  const float* row = P + (size_t)i * B_SZ;
  int li = labels[i];
  float m = -1e30f;
  for (int j = threadIdx.x; j < B_SZ; j += blockDim.x) m = fmaxf(m, row[j]);
  m = block_max(m, sh);
  float se = 0.f;
  for (int j = threadIdx.x; j < B_SZ; j += blockDim.x) se += __expf(row[j] - m);
  se = block_sum(se, sh);
  float lse = logf(se);
  float acc = 0.f;
  for (int j = threadIdx.x; j < B_SZ; j += blockDim.x) {
    float x = row[j] - m;
    float p  = __expf(x) / se;
    float lp = x - lse;
    float mn = (labels[j] == li) ? rsqrtf(cnt[j]) : 0.f;
    acc += p * (lp - logf(mn + 1e-8f));
  }
  acc = block_sum(acc, sh);
  if (threadIdx.x == 0) atomicAdd(accum, acc);
}

// ---------------- final assembly of the 5 losses + 2 precisions ------------------------
__global__ void finalize_kernel(const float* __restrict__ SE, const float* __restrict__ LB,
                                const unsigned long long* __restrict__ AK,
                                const int* __restrict__ labels, const float* __restrict__ SCAL,
                                const float* __restrict__ rowmax_s,
                                const float* __restrict__ colmax_im, float* __restrict__ out) {
  __shared__ float sh[256];
  float ce0 = 0.f, ce1 = 0.f, ce2 = 0.f, ce3 = 0.f, pv = 0.f, pt = 0.f, m1 = 0.f, m2 = 0.f;
  for (int i = threadIdx.x; i < B_SZ; i += blockDim.x) {
    ce0 += SHIFT_INST + logf(SE[i])            - LB[i];
    ce1 += SHIFT_INST + logf(SE[B_SZ + i])     - LB[B_SZ + i];
    ce2 += SHIFT_CMPC + logf(SE[2 * B_SZ + i]) - LB[2 * B_SZ + i];
    ce3 += SHIFT_CMPC + logf(SE[3 * B_SZ + i]) - LB[3 * B_SZ + i];
    pv += ((int)(unsigned int)(AK[i]        & 0xffffffffull) == labels[i]) ? 1.f : 0.f;
    pt += ((int)(unsigned int)(AK[B_SZ + i] & 0xffffffffull) == labels[i]) ? 1.f : 0.f;
    m1 += rowmax_s[i];
    m2 += colmax_im[i];
  }
  ce0 = block_sum(ce0, sh); ce1 = block_sum(ce1, sh);
  ce2 = block_sum(ce2, sh); ce3 = block_sum(ce3, sh);
  pv  = block_sum(pv,  sh); pt  = block_sum(pt,  sh);
  m1  = block_sum(m1,  sh); m2  = block_sum(m2,  sh);
  if (threadIdx.x == 0) {
    float invB = 1.0f / (float)B_SZ;
    out[0] = (ce0 + ce1) * invB;          // instance_loss
    out[1] = 2.0f * SCAL[0] * invB;       // global_align_loss
    out[2] = (ce2 + ce3) * invB;          // cmpc_loss
    out[3] = (SCAL[1] + SCAL[2]) * invB;  // cmpm_loss
    out[4] = m1 + m2;                     // mh_loss
    out[5] = pv * invB;                   // visual_prec
    out[6] = pt * invB;                   // textual_prec
  }
}

// ---------------- host-side orchestration ---------------------------------------------
extern "C" void kernel_launch(void* const* d_in, const int* in_sizes, int n_in,
                              void* d_out, int out_size, void* d_ws, size_t ws_size,
                              hipStream_t stream) {
  const float* V   = (const float*)d_in[0];
  const float* T   = (const float*)d_in[1];
  const float* Wp  = (const float*)d_in[2];
  const float* W2p = (const float*)d_in[3];
  const int*   lbl = (const int*)d_in[4];
  float* out = (float*)d_out;
  (void)in_sizes; (void)n_in; (void)out_size; (void)ws_size;

  char* w = (char*)d_ws;
  __bf16* vnb = (__bf16*)(w + (0u << 20));
  __bf16* tnb = (__bf16*)(w + (2u << 20));
  __bf16* vb  = (__bf16*)(w + (4u << 20));
  __bf16* tb  = (__bf16*)(w + (6u << 20));
  // 8MB..~29.5MB: transposed-weight slot (C_PAD x D bf16 = 21.5MB);
  // reused afterwards for sim/vp/tp (gemm_nt runs after all gemm_ce).
  __bf16* Wt  = (__bf16*)(w + (8u << 20));
  float*  sim = (float*)(w + (8u  << 20));
  float*  vp  = (float*)(w + (12u << 20));
  float*  tp  = (float*)(w + (16u << 20));
  // small region at 30MB: [zeroed: cnsqW, cnsqW2, SE, LB, AK, SCAL] then the rest
  float* cnsqW  = (float*)(w + (30u << 20));     // C_PAD
  float* cnsqW2 = cnsqW + C_PAD;                 // C_PAD
  float* SE     = cnsqW2 + C_PAD;                // 4*B  sumexp
  float* LB     = SE + 4 * B_SZ;                 // 4*B  label logits
  unsigned long long* AK = (unsigned long long*)(LB + 4 * B_SZ); // 2*B argmax keys
  float* SCAL   = (float*)(AK + 2 * B_SZ);       // [0]=ga [1]=cmpm_v [2]=cmpm_t (+pad)
  float* arow      = SCAL + 16;
  float* brow      = arow + B_SZ;
  float* rowmax_s  = brow + B_SZ;
  float* colmax_im = rowmax_s + B_SZ;
  float* cnt       = colmax_im + B_SZ;

  int zero_words = 2 * C_PAD + 4 * B_SZ + 4 * B_SZ + 4 * B_SZ /*AK as u32*/ + 16;
  zero_u32_kernel<<<(zero_words + 255) / 256, 256, 0, stream>>>((unsigned int*)cnsqW,
                                                                zero_words);

  prep_kernel<<<B_SZ, 256, 0, stream>>>(V, T, vnb, tnb, vb, tb, arow, brow);

  dim3 gct(C_PAD / 32, D_SZ / 32);
  dim3 gce(C_PAD / 128, B_SZ / 16);

  // ---- W: transpose+convert (+colnorm^2), then both instance CE GEMMs ----
  convT_kernel<<<gct, 256, 0, stream>>>(Wp, Wt, cnsqW);
  gemm_ce_kernel<<<gce, 256, 0, stream>>>(vnb, Wt, cnsqW, nullptr, 28.0f, SHIFT_INST,
                                          SE + 0 * B_SZ, LB + 0 * B_SZ, nullptr, lbl);
  gemm_ce_kernel<<<gce, 256, 0, stream>>>(tnb, Wt, cnsqW, nullptr, 28.0f, SHIFT_INST,
                                          SE + 1 * B_SZ, LB + 1 * B_SZ, nullptr, lbl);

  // ---- W2: reuse the same slot, then both cmpc CE GEMMs ----
  convT_kernel<<<gct, 256, 0, stream>>>(W2p, Wt, cnsqW2);
  gemm_ce_kernel<<<gce, 256, 0, stream>>>(tnb, Wt, cnsqW2, arow, 0.0f, SHIFT_CMPC,
                                          SE + 2 * B_SZ, LB + 2 * B_SZ, AK + 0 * B_SZ, lbl);
  gemm_ce_kernel<<<gce, 256, 0, stream>>>(vnb, Wt, cnsqW2, brow, 0.0f, SHIFT_CMPC,
                                          SE + 3 * B_SZ, LB + 3 * B_SZ, AK + 1 * B_SZ, lbl);

  // ---- B x B GEMMs (overwrite the weight slot; all gemm_ce are done) ----
  dim3 gnt(B_SZ / 128, B_SZ / 16);
  gemm_nt_kernel<<<gnt, 256, 0, stream>>>(vnb, tnb, sim);   // sim = vn @ tn^T
  gemm_nt_kernel<<<gnt, 256, 0, stream>>>(vb,  tnb, vp);    // vp  = v  @ tn^T
  gemm_nt_kernel<<<gnt, 256, 0, stream>>>(tb,  vnb, tp);    // tp  = t  @ vn^T

  lcnt_kernel<<<(B_SZ + 255) / 256, 256, 0, stream>>>(lbl, cnt);
  ga_mh_row_kernel<<<B_SZ, 256, 0, stream>>>(sim, lbl, &SCAL[0], rowmax_s);
  colmax_kernel<<<(B_SZ + 255) / 256, 256, 0, stream>>>(sim, lbl, colmax_im);
  cmpm_row_kernel<<<B_SZ, 256, 0, stream>>>(vp, lbl, cnt, &SCAL[1]);
  cmpm_row_kernel<<<B_SZ, 256, 0, stream>>>(tp, lbl, cnt, &SCAL[2]);

  finalize_kernel<<<1, 256, 0, stream>>>(SE, LB, AK, lbl, SCAL, rowmax_s, colmax_im, out);
}
